// DKN_61735859913344
// MI455X (gfx1250) — matrix operand order repaired and checked by reading, not compiled, not run on hardware
//
#include <hip/hip_runtime.h>
#include <hip/hip_bf16.h>
#include <cstdint>
#include <cstddef>

typedef __attribute__((ext_vector_type(16))) _Float16 v16h;
typedef __attribute__((ext_vector_type(8)))  _Float16 v8h;
typedef __attribute__((ext_vector_type(8)))  float    v8f;
typedef __attribute__((ext_vector_type(4)))  float    v4f;

// ---------------------------------------------------------------------------
// Bilinear upsample lr (2,1,64,64) -> depth (2,1,256,256), half-pixel centers
// ---------------------------------------------------------------------------
__global__ void dkn_upsample(const float* __restrict__ lr, float* __restrict__ depth) {
    int idx = blockIdx.x * blockDim.x + threadIdx.x;
    if (idx >= 2 * 256 * 256) return;
    int b = idx >> 16;
    int p = idx & 65535;
    int y = p >> 8, x = p & 255;
    float sy = (y + 0.5f) * 0.25f - 0.5f;
    float sx = (x + 0.5f) * 0.25f - 0.5f;
    float fy = floorf(sy), fx = floorf(sx);
    float wy = sy - fy, wx = sx - fx;
    int y0 = (int)fy, x0 = (int)fx;
    int y1 = y0 + 1, x1 = x0 + 1;
    y0 = min(max(y0, 0), 63); y1 = min(max(y1, 0), 63);
    x0 = min(max(x0, 0), 63); x1 = min(max(x1, 0), 63);
    const float* s = lr + b * 4096;
    float v = s[y0 * 64 + x0] * (1.f - wy) * (1.f - wx) +
              s[y0 * 64 + x1] * (1.f - wy) * wx +
              s[y1 * 64 + x0] * wy * (1.f - wx) +
              s[y1 * 64 + x1] * wy * wx;
    depth[idx] = v;
}

// ---------------------------------------------------------------------------
// Zero-pad (B,C,256,256) f32 NCHW -> (B,303,303,32) f16 NHWC (channel-padded)
// ---------------------------------------------------------------------------
__global__ void dkn_pad(const float* __restrict__ src, _Float16* __restrict__ dst,
                        int C, int padT, int padL) {
    int total = 2 * 303 * 303 * 32;
    for (int t = blockIdx.x * blockDim.x + threadIdx.x; t < total;
         t += gridDim.x * blockDim.x) {
        int c = t & 31;
        int r = t >> 5;
        int x = r % 303; r /= 303;
        int y = r % 303;
        int b = r / 303;
        int sy = y - padT, sx = x - padL;
        float v = 0.f;
        if (c < C && sy >= 0 && sy < 256 && sx >= 0 && sx < 256)
            v = src[(((size_t)b * C + c) * 256 + sy) * 256 + sx];
        dst[t] = (_Float16)v;
    }
}

// ---------------------------------------------------------------------------
// Fold BN into weights, convert to f16, pre-swizzle into a contiguous-k WMMA
// 16-bit A-matrix 16x32 lane layout.  GEMM-K order is (kh, kw, ci) over the
// channel-PADDED input (Cpad % 32 == 0), so each lane's A fragment and the
// matching NHWC B fragment are both contiguous 32-byte vectors.
// Packed layout: [mtile][kchunk][lane(32)][elem(16)]
// elem i of lane (half = lane>>4): K_in_chunk = half*16+i
// ---------------------------------------------------------------------------
__global__ void dkn_repack(const float* __restrict__ Wf, const float* __restrict__ bias,
                           const float* __restrict__ bng, const float* __restrict__ bnb,
                           const float* __restrict__ bnm, const float* __restrict__ bnv,
                           int has_bn, _Float16* __restrict__ Wp, float* __restrict__ bf,
                           int Cout, int Cin, int Cpad, int KH, int KW) {
    int nk = (KH * KW * Cpad) >> 5;
    int mt = Cout >> 4;
    int total = mt * nk * 512;
    for (int t = blockIdx.x * blockDim.x + threadIdx.x; t < total;
         t += gridDim.x * blockDim.x) {
        int i = t & 15;
        int lane = (t >> 4) & 31;
        int kc = (t >> 9) % nk;
        int mtile = t / (nk << 9);
        int half = lane >> 4;
        int M = mtile * 16 + (lane & 15);
        int kin = half * 16 + i;
        int K = kc * 32 + kin;             // (kh, kw, ci) order over Cpad
        int ci = K % Cpad;
        int cell = K / Cpad;
        int kh = cell / KW;
        int kw = cell % KW;
        float v = 0.f;
        if (ci < Cin) {
            float sc = 1.f;
            if (has_bn) sc = bng[M] * rsqrtf(bnv[M] + 1e-5f);
            v = Wf[(((size_t)M * Cin + ci) * KH + kh) * KW + kw] * sc;
        }
        Wp[t] = (_Float16)v;
    }
    for (int c = blockIdx.x * blockDim.x + threadIdx.x; c < Cout;
         c += gridDim.x * blockDim.x) {
        float bb = bias[c];
        if (has_bn) {
            float sc = bng[c] * rsqrtf(bnv[c] + 1e-5f);
            bb = (bb - bnm[c]) * sc + bnb[c];
        }
        bf[c] = bb;
    }
}

// ---------------------------------------------------------------------------
// Implicit-GEMM VALID convolution, NHWC activations, fused bias+ReLU.
// One wave computes a 16(Cout) x 64(spatial) macro-tile: one pre-swizzled A
// fragment feeds 4 WMMAs on 4 spatial 16-columns.  The whole A stream for
// this workgroup's mtile (nk x 1KB <= 50KB) is staged once into LDS by all
// 256 threads, then read back with ds_load_b128 -- all 8 waves share it, so
// global A traffic drops 8x and VMEM slots are left for the B stream.
// B fragments are contiguous 32B NHWC channel vectors (lane: col = lane&15,
// K = (lane>>4)*16 + i -> 16 consecutive channels at one input pixel).
// Epilogue: 8 consecutive output channels per lane -> one 16B store.
// EXEC is all-ones at every WMMA; edge tiles clamp addresses, predicate store.
// ---------------------------------------------------------------------------
__global__ __launch_bounds__(256) void dkn_conv_wmma(
    const _Float16* __restrict__ X, const _Float16* __restrict__ Wp,
    const float* __restrict__ bf, _Float16* __restrict__ Y,
    int Cpad, int Cout, int KH, int KW, int stride,
    int IH, int IW, int OH, int OW) {
    extern __shared__ _Float16 lds_a[];

    int wave = threadIdx.x >> 5;
    int lane = threadIdx.x & 31;
    int half = lane >> 4;
    int col  = lane & 15;
    int mtile = blockIdx.y;
    int b = blockIdx.z;
    int OHW = OH * OW;
    int nk = (KH * KW * Cpad) >> 5;

    // ---- cooperative stage of this mtile's packed A stream into LDS ----
    {
        const v4f* src = (const v4f*)(Wp + (size_t)mtile * nk * 512);
        v4f* dst = (v4f*)lds_a;
        int n16 = nk * 64;               // nk * 1024B / 16B
        for (int t = threadIdx.x; t < n16; t += 256) dst[t] = src[t];
    }
    __syncthreads();

    int tile0 = (blockIdx.x * 8 + wave) * 4;
    int pj[4];
    bool pv[4];
    const _Float16* xp[4];
#pragma unroll
    for (int j = 0; j < 4; ++j) {
        int pos = (tile0 + j) * 16 + col;
        pv[j] = pos < OHW;
        int p = pv[j] ? pos : (OHW - 1);
        pj[j] = p;
        int oh = p / OW;
        int ow = p - oh * OW;
        xp[j] = X + (((size_t)b * IH + oh * stride) * IW + ow * stride) * Cpad +
                (half << 4);
    }

    const _Float16* wa = lds_a + (lane << 4);
    v8f acc0 = {}, acc1 = {}, acc2 = {}, acc3 = {};
    for (int kh = 0; kh < KH; ++kh) {
        for (int kw = 0; kw < KW; ++kw) {
            size_t soff = ((size_t)kh * IW + kw) * Cpad;
            for (int cb = 0; cb < Cpad; cb += 32) {
                v16h a = *(const v16h*)wa;   // ds_load_b128 x2
                wa += 512;
                size_t o = soff + cb;
                v16h b0 = *(const v16h*)(xp[0] + o);
                v16h b1 = *(const v16h*)(xp[1] + o);
                v16h b2 = *(const v16h*)(xp[2] + o);
                v16h b3 = *(const v16h*)(xp[3] + o);
                acc0 = __builtin_amdgcn_wmma_f32_16x16x32_f16(
                    false, a, false, b0, (short)0, acc0, false, false);
                acc1 = __builtin_amdgcn_wmma_f32_16x16x32_f16(
                    false, a, false, b1, (short)0, acc1, false, false);
                acc2 = __builtin_amdgcn_wmma_f32_16x16x32_f16(
                    false, a, false, b2, (short)0, acc2, false, false);
                acc3 = __builtin_amdgcn_wmma_f32_16x16x32_f16(
                    false, a, false, b3, (short)0, acc3, false, false);
            }
        }
    }

    int cb8 = mtile * 16 + half * 8;   // 8 consecutive output channels
    float bb[8];
#pragma unroll
    for (int r = 0; r < 8; ++r) bb[r] = bf[cb8 + r];

    v8f* accs[4] = {&acc0, &acc1, &acc2, &acc3};
#pragma unroll
    for (int j = 0; j < 4; ++j) {
        if (!pv[j]) continue;
        v8h ov;
#pragma unroll
        for (int r = 0; r < 8; ++r) {
            float v = (*accs[j])[r] + bb[r];
            v = v > 0.f ? v : 0.f;
            ov[r] = (_Float16)v;
        }
        *(v8h*)(Y + ((size_t)b * OHW + pj[j]) * Cout + cb8) = ov;
    }
}

// ---------------------------------------------------------------------------
// Head: 1x1 convs 128 -> 9 (sigmoid weight) and 128 -> 18 (offset) on NHWC
// activations, with shift-and-stitch scatter into full-res buffers.
// ---------------------------------------------------------------------------
__global__ void dkn_head(const _Float16* __restrict__ X,
                         const float* __restrict__ Ww, const float* __restrict__ Bw,
                         const float* __restrict__ Wo, const float* __restrict__ Bo,
                         float* __restrict__ Wst, float* __restrict__ Ost,
                         int si, int sj) {
    int idx = blockIdx.x * blockDim.x + threadIdx.x;
    if (idx >= 2 * 64 * 64) return;
    int b = idx >> 12;
    int p = idx & 4095;
    int h = p >> 6, w = p & 63;
    const _Float16* xp = X + ((size_t)b * 4096 + p) * 128;  // NHWC: contiguous
    float aw[9];
    float ao[18];
#pragma unroll
    for (int c = 0; c < 9; ++c) aw[c] = 0.f;
#pragma unroll
    for (int c = 0; c < 18; ++c) ao[c] = 0.f;
    for (int k = 0; k < 128; ++k) {
        float xv = (float)xp[k];
#pragma unroll
        for (int c = 0; c < 9; ++c) aw[c] += xv * Ww[c * 128 + k];
#pragma unroll
        for (int c = 0; c < 18; ++c) ao[c] += xv * Wo[c * 128 + k];
    }
    int H = h * 4 + si;
    int W2 = w * 4 + sj;
    size_t o = (size_t)H * 256 + W2;
#pragma unroll
    for (int c = 0; c < 9; ++c) {
        float v = aw[c] + Bw[c];
        v = 1.f / (1.f + expf(-v));
        Wst[((size_t)b * 9 + c) * 65536 + o] = v;
    }
#pragma unroll
    for (int c = 0; c < 18; ++c)
        Ost[((size_t)b * 18 + c) * 65536 + o] = ao[c] + Bo[c];
}

// ---------------------------------------------------------------------------
// Combine the two networks, subtract channel-mean of weights, deformable
// 3x3 bilinear sampling on the (virtual) 15x15 zero-padded depth patch,
// weighted sum + residual.
// ---------------------------------------------------------------------------
__global__ void dkn_output(const float* __restrict__ Wi, const float* __restrict__ Oi,
                           const float* __restrict__ Wd, const float* __restrict__ Od,
                           const float* __restrict__ depth, float* __restrict__ out) {
    int idx = blockIdx.x * blockDim.x + threadIdx.x;
    if (idx >= 2 * 256 * 256) return;
    int b = idx >> 16;
    int p = idx & 65535;
    int Y = p >> 8, X = p & 255;
    size_t bw = (size_t)b * 9 * 65536;
    size_t bo = (size_t)b * 18 * 65536;
    float w[9];
    float sum = 0.f;
#pragma unroll
    for (int c = 0; c < 9; ++c) {
        w[c] = Wi[bw + (size_t)c * 65536 + p] * Wd[bw + (size_t)c * 65536 + p];
        sum += w[c];
    }
    float mean = sum * (1.f / 9.f);
    const float* dp = depth + (size_t)b * 65536;
    float acc = 0.f;
    for (int ti = 0; ti < 3; ++ti) {
        for (int tj = 0; tj < 3; ++tj) {
            int t = ti * 3 + tj;
            float ox = Oi[bo + (size_t)(2 * t) * 65536 + p] *
                       Od[bo + (size_t)(2 * t) * 65536 + p];
            float oy = Oi[bo + (size_t)(2 * t + 1) * 65536 + p] *
                       Od[bo + (size_t)(2 * t + 1) * 65536 + p];
            // coord -> patch coords simplifies to grid + offset - 0.5
            float xs = (float)(7 + ti) + ox - 0.5f;
            float ys = (float)(7 + tj) + oy - 0.5f;
            float xf = floorf(xs), yf = floorf(ys);
            float wx = xs - xf, wy = ys - yf;
            int x0 = (int)xf, y0 = (int)yf;
            float v00 = 0.f, v01 = 0.f, v10 = 0.f, v11 = 0.f;
#pragma unroll
            for (int dy = 0; dy < 2; ++dy) {
#pragma unroll
                for (int dx = 0; dx < 2; ++dx) {
                    int py = y0 + dy, px = x0 + dx;
                    float v = 0.f;
                    if (py >= 0 && py < 15 && px >= 0 && px < 15) {
                        int du = Y + py - 7, dv = X + px - 7;
                        if (du >= 0 && du < 256 && dv >= 0 && dv < 256)
                            v = dp[du * 256 + dv];
                    }
                    if (dy == 0 && dx == 0) v00 = v;
                    else if (dy == 0) v01 = v;
                    else if (dx == 0) v10 = v;
                    else v11 = v;
                }
            }
            float s = v00 * (1.f - wx) * (1.f - wy) + v01 * wx * (1.f - wy) +
                      v10 * (1.f - wx) * wy + v11 * wx * wy;
            acc += s * (w[t] - mean);
        }
    }
    out[idx] = acc + dp[p];
}

// ---------------------------------------------------------------------------
// Host driver
// ---------------------------------------------------------------------------
struct ConvSpec {
    int Cin, Cpad, Cout, KH, KW, st, IH, IW, OH, OW, wi, bi, bni;
};

extern "C" void kernel_launch(void* const* d_in, const int* in_sizes, int n_in,
                              void* d_out, int out_size, void* d_ws, size_t ws_size,
                              hipStream_t stream) {
    (void)in_sizes; (void)n_in; (void)out_size; (void)ws_size;
    const float* lr  = (const float*)d_in[0];
    const float* rgb = (const float*)d_in[1];
    const int BASE[2] = {2, 32};  // im_params at d_in[2..31], dp_params at [32..61]
    auto PP = [&](int net, int i) -> const float* {
        return (const float*)d_in[BASE[net] + i];
    };

    char* wsb = (char*)d_ws;
    size_t off = 0;
    auto alloc = [&](size_t bytes) -> void* {
        off = (off + 255) & ~(size_t)255;
        void* ptr = wsb + off;
        off += bytes;
        return ptr;
    };

    float*    depth = (float*)alloc((size_t)2 * 65536 * 4);
    _Float16* x0    = (_Float16*)alloc((size_t)2 * 303 * 303 * 32 * 2);  // NHWC
    _Float16* act[8];
    act[1] = (_Float16*)alloc((size_t)2 * 297 * 297 * 32 * 2);
    act[2] = (_Float16*)alloc((size_t)2 * 148 * 148 * 32 * 2);
    act[3] = (_Float16*)alloc((size_t)2 * 144 * 144 * 64 * 2);
    act[4] = (_Float16*)alloc((size_t)2 * 72 * 72 * 64 * 2);
    act[5] = (_Float16*)alloc((size_t)2 * 68 * 68 * 128 * 2);
    act[6] = (_Float16*)alloc((size_t)2 * 66 * 66 * 128 * 2);
    act[7] = (_Float16*)alloc((size_t)2 * 64 * 64 * 128 * 2);
    float* Wst[2]; float* Ost[2];
    for (int n = 0; n < 2; ++n) {
        Wst[n] = (float*)alloc((size_t)2 * 9 * 65536 * 4);
        Ost[n] = (float*)alloc((size_t)2 * 18 * 65536 * 4);
    }

    // Layer specs (param indices follow JAX pytree sorted-key flattening:
    // b1..b7=0..6, bn1=7..10{b,g,m,v}, bn3=11..14, bn5=15..18, bo=19, bw=20,
    // w1..w7=21..27, wo=28, ww=29)
    ConvSpec specs[2][7];
    for (int n = 0; n < 2; ++n) {
        int cin = (n == 0) ? 3 : 1;
        ConvSpec s[7] = {
            {cin, 32, 32, 7, 7, 1, 303, 303, 297, 297, 21, 0, 7},
            {32, 32, 32, 2, 2, 2, 297, 297, 148, 148, 22, 1, -1},
            {32, 32, 64, 5, 5, 1, 148, 148, 144, 144, 23, 2, 11},
            {64, 64, 64, 2, 2, 2, 144, 144, 72, 72, 24, 3, -1},
            {64, 64, 128, 5, 5, 1, 72, 72, 68, 68, 25, 4, 15},
            {128, 128, 128, 3, 3, 1, 68, 68, 66, 66, 26, 5, -1},
            {128, 128, 128, 3, 3, 1, 66, 66, 64, 64, 27, 6, -1}};
        for (int l = 0; l < 7; ++l) specs[n][l] = s[l];
    }

    // ---- repack all conv weights once (BN folded, WMMA A-layout, f16) ----
    _Float16* wp[2][7];
    float*    bfold[2][7];
    for (int n = 0; n < 2; ++n) {
        for (int l = 0; l < 7; ++l) {
            const ConvSpec& s = specs[n][l];
            int nk = (s.KH * s.KW * s.Cpad) >> 5;
            int mt = s.Cout >> 4;
            int total = mt * nk * 512;
            wp[n][l]    = (_Float16*)alloc((size_t)total * 2);
            bfold[n][l] = (float*)alloc((size_t)s.Cout * 4);
            const float* bng = nullptr; const float* bnb = nullptr;
            const float* bnm = nullptr; const float* bnv = nullptr;
            int hb = 0;
            if (s.bni >= 0) {
                hb = 1;
                bnb = PP(n, s.bni + 0);
                bng = PP(n, s.bni + 1);
                bnm = PP(n, s.bni + 2);
                bnv = PP(n, s.bni + 3);
            }
            dkn_repack<<<(total + 255) / 256, 256, 0, stream>>>(
                PP(n, s.wi), PP(n, s.bi), bng, bnb, bnm, bnv, hb,
                wp[n][l], bfold[n][l], s.Cout, s.Cin, s.Cpad, s.KH, s.KW);
        }
    }

    // ---- depth upsample ----
    dkn_upsample<<<(2 * 65536 + 255) / 256, 256, 0, stream>>>(lr, depth);

    auto runconv = [&](const _Float16* X, const ConvSpec& s,
                       const _Float16* Wp2, const float* bf2, _Float16* Y) {
        int OHW = s.OH * s.OW;
        int ntiles = (OHW + 15) / 16;          // 16-wide spatial tiles
        int nk = (s.KH * s.KW * s.Cpad) >> 5;
        size_t shmem = (size_t)nk * 1024;      // staged A stream (<= 50KB)
        dim3 g((ntiles + 31) / 32, s.Cout / 16, 2);  // 8 waves x 4 tiles/block
        dkn_conv_wmma<<<g, 256, shmem, stream>>>(X, Wp2, bf2, Y, s.Cpad, s.Cout,
                                                 s.KH, s.KW, s.st, s.IH, s.IW,
                                                 s.OH, s.OW);
    };

    // ---- 16 shift positions x {image, depth} networks ----
    for (int si = 0; si < 4; ++si) {
        for (int sj = 0; sj < 4; ++sj) {
            int padT = 25 - si, padL = 25 - sj;
            for (int n = 0; n < 2; ++n) {
                const float* src = (n == 0) ? rgb : depth;
                int C = (n == 0) ? 3 : 1;
                int ptotal = 2 * 303 * 303 * 32;
                dkn_pad<<<(ptotal + 255) / 256, 256, 0, stream>>>(src, x0, C,
                                                                  padT, padL);
                const _Float16* xin = x0;
                for (int l = 0; l < 7; ++l) {
                    runconv(xin, specs[n][l], wp[n][l], bfold[n][l], act[l + 1]);
                    xin = act[l + 1];
                }
                dkn_head<<<64, 128, 0, stream>>>(act[7],
                                                 PP(n, 29), PP(n, 20),  // ww, bw
                                                 PP(n, 28), PP(n, 19),  // wo, bo
                                                 Wst[n], Ost[n], si, sj);
            }
        }
    }

    // ---- combine + deformable sampling + residual ----
    dkn_output<<<(2 * 65536 + 255) / 256, 256, 0, stream>>>(
        Wst[0], Ost[0], Wst[1], Ost[1], depth, (float*)d_out);
}
